// Spike_PointCNN_84791244357637
// MI455X (gfx1250) — compile-verified, matrix operand order checked by reading, not run on hardware
//
#include <hip/hip_runtime.h>
#include <math.h>
#include <stdint.h>

// ---------------------------------------------------------------------------
// Spike-PointCNN forward for MI455X (gfx1250, wave32).
// Dense layers -> v_wmma_f32_16x16x32_bf16 (f32 accumulate), B operand staged
// through LDS via global_load_async_to_lds_b128 (ASYNCcnt path).
// Irregular work (knn/fps/BN-stats/X-transform) -> VALU/LDS kernels.
// ---------------------------------------------------------------------------

#define BB 32
#define VTH 1.0f
#define BN_EPS 1e-5f

typedef __attribute__((ext_vector_type(16))) __bf16 v16bf;
typedef __attribute__((ext_vector_type(8)))  float  v8f;

static __device__ __forceinline__ __bf16 f2bf(float f) {
  union { float f; unsigned u; } v; v.f = f;
  unsigned r = v.u + 0x7FFFu + ((v.u >> 16) & 1u);   // round-to-nearest-even
  unsigned short h = (unsigned short)(r >> 16);
  return __builtin_bit_cast(__bf16, h);
}

// ---------------- generic GEMM: C[M,N] = A[M,Kd] * W[Kd,N] + bias ----------
// block = 4 waves, each wave owns a 16x16 tile of a 64(M) x 16(N) block tile.
// All waves share one LDS-staged B tile (32x16 f32) per K-step.
// N must be a multiple of 16 (true for every layer in this network).
__global__ void gemm_bf16_wmma(const float* __restrict__ A, const float* __restrict__ W,
                               const float* __restrict__ bias, float* __restrict__ Cmat,
                               int M, int N, int Kd) {
  __shared__ float Bs[32][16];          // 2 KB, b128-aligned rows

  const int lane = threadIdx.x;         // 0..31
  const int half = lane >> 4;           // 0/1
  const int l16  = lane & 15;
  const int tm   = blockIdx.y * blockDim.y + threadIdx.y;
  const int row0 = tm * 16;
  const int col0 = blockIdx.x * 16;

  // staging role: 128 threads cover 32 rows x 4 col-groups (float4 each)
  const int t    = threadIdx.y * 32 + threadIdx.x;   // 0..127
  const int srow = t >> 2;
  const int scol = (t & 3) << 2;

  const int  arow = row0 + l16;
  const bool arok = (arow < M);
  const int  arow_c = arok ? arow : (M - 1);          // clamp: garbage rows never stored
  const float* Arow = A + (size_t)arow_c * Kd;
  const int  bcol = col0 + l16;
  const bool bcok = (bcol < N);

  v8f acc = {};
  for (int k0 = 0; k0 < Kd; k0 += 32) {
    // ---- stage B tile into LDS via async DMA (coalesced b128 per thread) ----
    const int kr = k0 + srow;
    if (kr < Kd) {
      unsigned dst = (unsigned)(uintptr_t)(&Bs[srow][scol]);        // LDS byte offset
      unsigned long long src =
          (unsigned long long)(uintptr_t)(W + (size_t)kr * N + col0 + scol);
      asm volatile("global_load_async_to_lds_b128 %0, %1, off"
                   :: "v"(dst), "v"(src) : "memory");
    } else {
      *(float4*)(&Bs[srow][scol]) = make_float4(0.f, 0.f, 0.f, 0.f);
    }
    asm volatile("s_wait_asynccnt 0" ::: "memory");
    __syncthreads();

    if (k0 + 64 <= Kd) __builtin_prefetch(Arow + k0 + 32, 0, 3);    // global_prefetch_b8

    // ---- A fragment: elements e<8 -> k=k0+half*8+e ; e>=8 -> +16 ----
    v16bf a{};
    if (k0 + 32 <= Kd) {                       // fast path: 4x global_load_b128
      const float* p0 = Arow + k0 + half * 8;
      float4 f0 = *(const float4*)(p0);
      float4 f1 = *(const float4*)(p0 + 4);
      float4 f2 = *(const float4*)(p0 + 16);
      float4 f3 = *(const float4*)(p0 + 20);
      a[0] = f2bf(f0.x);  a[1] = f2bf(f0.y);  a[2] = f2bf(f0.z);  a[3] = f2bf(f0.w);
      a[4] = f2bf(f1.x);  a[5] = f2bf(f1.y);  a[6] = f2bf(f1.z);  a[7] = f2bf(f1.w);
      a[8] = f2bf(f2.x);  a[9] = f2bf(f2.y);  a[10] = f2bf(f2.z); a[11] = f2bf(f2.w);
      a[12] = f2bf(f3.x); a[13] = f2bf(f3.y); a[14] = f2bf(f3.z); a[15] = f2bf(f3.w);
    } else {                                   // K-remainder path (Kd=3/24 layers)
#pragma unroll
      for (int e = 0; e < 16; ++e) {
        int k = k0 + ((e >> 3) << 4) + half * 8 + (e & 7);
        a[e] = f2bf((k < Kd) ? Arow[k] : 0.0f);
      }
    }

    // ---- B fragment from LDS (conflict-free ds_load_b32) ----
    v16bf b{};
#pragma unroll
    for (int e = 0; e < 16; ++e) {
      int k = ((e >> 3) << 4) + half * 8 + (e & 7);
      b[e] = f2bf(Bs[k][l16]);
    }

    acc = __builtin_amdgcn_wmma_f32_16x16x32_bf16(false, a, false, b, (short)0, acc,
                                                  false, false);
    __syncthreads();                           // before next tile overwrites Bs
  }

  if (bcok) {
    const float bv = bias ? bias[bcol] : 0.0f;
#pragma unroll
    for (int v = 0; v < 8; ++v) {
      const int row = row0 + v + half * 8;     // C/D layout: VGPR v -> M = v + 8*half
      if (row < M) Cmat[(size_t)row * N + bcol] = acc[v] + bv;
    }
  }
}

// ---------------- elementwise / stats kernels ------------------------------
__global__ void zero_f32(float* p, int n) {
  int i = blockIdx.x * blockDim.x + threadIdx.x;
  if (i < n) p[i] = 0.0f;
}

__global__ void spike_inplace(float* z, long n) {
  long stride = (long)gridDim.x * blockDim.x;
  for (long i = (long)blockIdx.x * blockDim.x + threadIdx.x; i < n; i += stride)
    z[i] = (z[i] >= VTH) ? 1.0f : 0.0f;
}

// per-channel sum of values (channel-last layout [rows, C])
__global__ void colsum(const float* __restrict__ z, float* __restrict__ sum,
                       int rows, int C) {
  const int c = blockIdx.y;
  const int tid = threadIdx.x;
  float acc = 0.0f;
  for (int r = blockIdx.x * blockDim.x + tid; r < rows; r += gridDim.x * blockDim.x)
    acc += z[(size_t)r * C + c];
  __shared__ float red[256];
  red[tid] = acc; __syncthreads();
  for (int s = 128; s > 0; s >>= 1) { if (tid < s) red[tid] += red[tid + s]; __syncthreads(); }
  if (tid == 0) atomicAdd(&sum[c], red[0]);
}

__global__ void colsum2(const float* __restrict__ z, float* __restrict__ sum,
                        float* __restrict__ sumsq, int rows, int C) {
  const int c = blockIdx.y;
  const int tid = threadIdx.x;
  float a = 0.0f, a2 = 0.0f;
  for (int r = blockIdx.x * blockDim.x + tid; r < rows; r += gridDim.x * blockDim.x) {
    float v = z[(size_t)r * C + c];
    a += v; a2 += v * v;
  }
  __shared__ float r1[256], r2[256];
  r1[tid] = a; r2[tid] = a2; __syncthreads();
  for (int s = 128; s > 0; s >>= 1) {
    if (tid < s) { r1[tid] += r1[tid + s]; r2[tid] += r2[tid + s]; }
    __syncthreads();
  }
  if (tid == 0) { atomicAdd(&sum[c], r1[0]); atomicAdd(&sumsq[c], r2[0]); }
}

// for binary (post-spike) data: var = m - m^2
__global__ void prep_binary(const float* sum, const float* g, const float* be,
                            float* scale, float* shift, float inv_rows, int C) {
  int c = threadIdx.x;
  if (c < C) {
    float m  = sum[c] * inv_rows;
    float var = m - m * m;
    float sc = g[c] * rsqrtf(var + BN_EPS);
    scale[c] = sc;
    shift[c] = be[c] - m * sc;
  }
}

__global__ void prep_moments(const float* sum, const float* sumsq, const float* g,
                             const float* be, float* scale, float* shift,
                             float inv_rows, int C) {
  int c = threadIdx.x;
  if (c < C) {
    float m   = sum[c] * inv_rows;
    float var = sumsq[c] * inv_rows - m * m;
    float sc  = g[c] * rsqrtf(var + BN_EPS);
    scale[c]  = sc;
    shift[c]  = be[c] - m * sc;
  }
}

__global__ void affine_inplace(float* z, const float* scale, const float* shift,
                               long rows, int C) {
  long n = rows * C;
  long stride = (long)gridDim.x * blockDim.x;
  for (long i = (long)blockIdx.x * blockDim.x + threadIdx.x; i < n; i += stride) {
    int c = (int)(i % C);
    z[i] = z[i] * scale[c] + shift[c];
  }
}

// ---------------- geometry kernels -----------------------------------------
__global__ void compute_sq(const float* __restrict__ pos, float* __restrict__ sq, int n) {
  int i = blockIdx.x * blockDim.x + threadIdx.x;
  if (i < n) {
    float x = pos[3 * (size_t)i], y = pos[3 * (size_t)i + 1], z = pos[3 * (size_t)i + 2];
    sq[i] = x * x + y * y + z * z;
  }
}

// per query point: keep Kd smallest d2 (ascending, ties -> lower index),
// emit every dil-th -> Kout neighbors
__global__ void knn_kernel(const float* __restrict__ pos, const float* __restrict__ sq,
                           int* __restrict__ idx, int Pp, int Kd, int dil, int Kout) {
  int t = blockIdx.x * blockDim.x + threadIdx.x;
  if (t >= BB * Pp) return;
  int b = t / Pp, p = t % Pp;
  const float* pb  = pos + (size_t)b * Pp * 3;
  const float* sqb = sq  + (size_t)b * Pp;
  float px = pb[3 * p], py = pb[3 * p + 1], pz = pb[3 * p + 2];
  float sp = sqb[p];
  float nd[32]; int ni[32];
  for (int i = 0; i < Kd; i++) { nd[i] = 3.0e38f; ni[i] = 0; }
  for (int q = 0; q < Pp; q++) {
    float d = sp + sqb[q] - 2.0f * (px * pb[3 * q] + py * pb[3 * q + 1] + pz * pb[3 * q + 2]);
    if (d < nd[Kd - 1]) {
      int j = Kd - 1;
      while (j > 0 && nd[j - 1] > d) { nd[j] = nd[j - 1]; ni[j] = ni[j - 1]; j--; }
      nd[j] = d; ni[j] = q;
    }
  }
  for (int j = 0; j < Kout; j++) idx[(size_t)t * Kout + j] = ni[j * dil];
}

__global__ void rel_kernel(const float* __restrict__ pos, const int* __restrict__ idx,
                           float* __restrict__ rel, int Pp, int K) {
  long tot = (long)BB * Pp * K;
  long stride = (long)gridDim.x * blockDim.x;
  for (long i = (long)blockIdx.x * blockDim.x + threadIdx.x; i < tot; i += stride) {
    long r = i / K;
    int p = (int)(r % Pp), b = (int)(r / Pp);
    int q = idx[i];
    const float* pb = pos + (size_t)b * Pp * 3;
    rel[3 * i]     = pb[3 * q]     - pb[3 * p];
    rel[3 * i + 1] = pb[3 * q + 1] - pb[3 * p + 1];
    rel[3 * i + 2] = pb[3 * q + 2] - pb[3 * p + 2];
  }
}

// deterministic FPS starting at index 0; one block per batch element
__global__ void fps_kernel(const float* __restrict__ pos, int* __restrict__ out_idx,
                           int Pp, int n_s) {
  const int b = blockIdx.x, tid = threadIdx.x;
  __shared__ float dist[1024];
  __shared__ float redv[256];
  __shared__ int   redi[256];
  const float* pb = pos + (size_t)b * Pp * 3;
  float x0 = pb[0], y0 = pb[1], z0 = pb[2];
  for (int i = tid; i < Pp; i += 256) {
    float dx = pb[3 * i] - x0, dy = pb[3 * i + 1] - y0, dz = pb[3 * i + 2] - z0;
    dist[i] = dx * dx + dy * dy + dz * dz;
  }
  if (tid == 0) out_idx[(size_t)b * n_s] = 0;
  __syncthreads();
  for (int s = 1; s < n_s; s++) {
    float bv = -3.0e38f; int bi = 0x7fffffff;
    for (int i = tid; i < Pp; i += 256)
      if (dist[i] > bv) { bv = dist[i]; bi = i; }
    redv[tid] = bv; redi[tid] = bi; __syncthreads();
    for (int w = 128; w > 0; w >>= 1) {
      if (tid < w) {
        if (redv[tid + w] > redv[tid] ||
            (redv[tid + w] == redv[tid] && redi[tid + w] < redi[tid])) {
          redv[tid] = redv[tid + w]; redi[tid] = redi[tid + w];
        }
      }
      __syncthreads();
    }
    int sel = redi[0];
    if (tid == 0) out_idx[(size_t)b * n_s + s] = sel;
    float sx = pb[3 * sel], sy = pb[3 * sel + 1], sz = pb[3 * sel + 2];
    for (int i = tid; i < Pp; i += 256) {
      float dx = pb[3 * i] - sx, dy = pb[3 * i + 1] - sy, dz = pb[3 * i + 2] - sz;
      float nd = dx * dx + dy * dy + dz * dz;
      if (nd < dist[i]) dist[i] = nd;
    }
    __syncthreads();
  }
}

__global__ void gather_rows(const float* __restrict__ src, const int* __restrict__ idx,
                            float* __restrict__ dst, int Pin, int Pout, int C) {
  long tot = (long)BB * Pout * C;
  long stride = (long)gridDim.x * blockDim.x;
  for (long i = (long)blockIdx.x * blockDim.x + threadIdx.x; i < tot; i += stride) {
    int c = (int)(i % C);
    long r = i / C;
    int po = (int)(r % Pout), b = (int)(r / Pout);
    dst[i] = src[((size_t)b * Pin + idx[(size_t)b * Pout + po]) * C + c];
  }
}

// t2[np,g,j] = sum_k t1[np,g,k] * Wc[g,j,k] + bc[g*K+j]; optional spike
__global__ void xform_conv(const float* __restrict__ tin, const float* __restrict__ Wc,
                           const float* __restrict__ bc, float* __restrict__ tout,
                           int NP, int K, int do_spike) {
  long tot = (long)NP * K * K;
  long stride = (long)gridDim.x * blockDim.x;
  for (long i = (long)blockIdx.x * blockDim.x + threadIdx.x; i < tot; i += stride) {
    int j = (int)(i % K);
    long r = i / K;
    int g = (int)(r % K);
    long np = r / K;
    const float* trow = tin + (np * K + g) * K;
    const float* wrow = Wc + ((size_t)g * K + j) * K;
    float a = bc[g * K + j];
    for (int k = 0; k < K; k++) a += trow[k] * wrow[k];
    if (do_spike) a = (a >= VTH) ? 1.0f : 0.0f;
    tout[i] = a;
  }
}

// y[np, c*dm+m] = sum_k ( sum_k2 hc[np,k2,c] * t[np,k2,k] ) * Wd[c,m,k] + bd
// hc is concat([h (Cd channels), xprev gathered via idx (Cin channels)]) -- fused
__global__ void depthwise_kernel(const float* __restrict__ h, const float* __restrict__ xprev,
                                 const int* __restrict__ idx, const float* __restrict__ t,
                                 const float* __restrict__ Wd, const float* __restrict__ bd,
                                 float* __restrict__ y, int NP, int Pp, int C, int Cd,
                                 int Cin, int K, int dm) {
  long tot = (long)NP * C * dm;
  long stride = (long)gridDim.x * blockDim.x;
  for (long i = (long)blockIdx.x * blockDim.x + threadIdx.x; i < tot; i += stride) {
    int m = (int)(i % dm);
    long r = i / dm;
    int c = (int)(r % C);
    long np = r / C;
    int b = (int)(np / Pp);
    float xt[16];
    for (int k = 0; k < K; k++) xt[k] = 0.0f;
    for (int k2 = 0; k2 < K; k2++) {
      float hv;
      if (c < Cd) hv = h[(np * K + k2) * Cd + c];
      else {
        int q = idx[np * K + k2];
        hv = xprev[((size_t)b * Pp + q) * Cin + (c - Cd)];
      }
      const float* trow = t + (np * K + k2) * K;
      for (int k = 0; k < K; k++) xt[k] += hv * trow[k];
    }
    float acc = 0.0f;
    const float* wrow = Wd + ((size_t)c * dm + m) * K;
    for (int k = 0; k < K; k++) acc += xt[k] * wrow[k];
    y[np * (C * dm) + c * dm + m] = acc + bd[c * dm + m];
  }
}

// mean-pool over points + 3-layer MLP head + log_softmax; one block per batch
__global__ void head_kernel(const float* __restrict__ x, int Pp,
                            const float* __restrict__ Wl1, const float* __restrict__ bl1,
                            const float* __restrict__ Wl2, const float* __restrict__ bl2,
                            const float* __restrict__ Wl3, const float* __restrict__ bl3,
                            float* __restrict__ out) {
  const int b = blockIdx.x, tid = threadIdx.x;
  __shared__ float g[384], h1[256], h2[128], o[40];
  for (int c = tid; c < 384; c += blockDim.x) {
    float s = 0.0f;
    for (int p = 0; p < Pp; p++) s += x[((size_t)b * Pp + p) * 384 + c];
    g[c] = s / (float)Pp;
  }
  __syncthreads();
  for (int j = tid; j < 256; j += blockDim.x) {
    float a = bl1[j];
    for (int c = 0; c < 384; c++) a += g[c] * Wl1[(size_t)c * 256 + j];
    h1[j] = (a >= VTH) ? 1.0f : 0.0f;
  }
  __syncthreads();
  for (int j = tid; j < 128; j += blockDim.x) {
    float a = bl2[j];
    for (int c = 0; c < 256; c++) a += h1[c] * Wl2[(size_t)c * 128 + j];
    h2[j] = (a >= VTH) ? 1.0f : 0.0f;
  }
  __syncthreads();
  for (int j = tid; j < 40; j += blockDim.x) {
    float a = bl3[j];
    for (int c = 0; c < 128; c++) a += h2[c] * Wl3[(size_t)c * 40 + j];
    o[j] = a;
  }
  __syncthreads();
  if (tid == 0) {
    float mx = o[0];
    for (int j = 1; j < 40; j++) mx = fmaxf(mx, o[j]);
    float s = 0.0f;
    for (int j = 0; j < 40; j++) s += expf(o[j] - mx);
    float ls = logf(s);
    for (int j = 0; j < 40; j++) out[(size_t)b * 40 + j] = o[j] - mx - ls;
  }
}

// ---------------------------------------------------------------------------
// host-side orchestration
// ---------------------------------------------------------------------------
static inline int ceil_div(int a, int b) { return (a + b - 1) / b; }
static inline dim3 gs_grid(long n) {
  long bl = (n + 255) / 256;
  if (bl > 16384) bl = 16384;
  if (bl < 1) bl = 1;
  return dim3((unsigned)bl);
}

static void launch_gemm(hipStream_t s, const float* A, const float* W, const float* bias,
                        float* Cmat, int M, int N, int Kd) {
  dim3 blk(32, 4);
  dim3 grd(ceil_div(N, 16), ceil_div(M, 64));
  gemm_bf16_wmma<<<grd, blk, 0, s>>>(A, W, bias, Cmat, M, N, Kd);
}

static void bn_binary(hipStream_t s, float* z, int rows, int C, const float* g,
                      const float* be, float* sum, float* scale, float* shift) {
  zero_f32<<<1, 256, 0, s>>>(sum, C);
  colsum<<<dim3(8, C), 256, 0, s>>>(z, sum, rows, C);
  prep_binary<<<1, 256, 0, s>>>(sum, g, be, scale, shift, 1.0f / rows, C);
  affine_inplace<<<gs_grid((long)rows * C), 256, 0, s>>>(z, scale, shift, rows, C);
}

static void bn_spike(hipStream_t s, float* z, int rows, int C, const float* g,
                     const float* be, float* sum, float* scale, float* shift) {
  spike_inplace<<<gs_grid((long)rows * C), 256, 0, s>>>(z, (long)rows * C);
  bn_binary(s, z, rows, C, g, be, sum, scale, shift);
}

static void bn_moments(hipStream_t s, float* z, int rows, int C, const float* g,
                       const float* be, float* sum, float* sumsq, float* scale,
                       float* shift) {
  zero_f32<<<1, 256, 0, s>>>(sum, C);
  zero_f32<<<1, 256, 0, s>>>(sumsq, C);
  colsum2<<<dim3(8, C), 256, 0, s>>>(z, sum, sumsq, rows, C);
  prep_moments<<<1, 256, 0, s>>>(sum, sumsq, g, be, scale, shift, 1.0f / rows, C);
  affine_inplace<<<gs_grid((long)rows * C), 256, 0, s>>>(z, scale, shift, rows, C);
}

struct StageCfg { int Cin, Cout, Cd, K, dil; };

// prm order (per-stage, insertion order of the reference dict):
// 0:W1 1:b1 2:g1 3:be1 4:W2 5:b2 6:g2 7:be2 8:Wm 9:bm 10:gm 11:bem
// 12:Wc1 13:bc1 14:gc1 15:bec1 16:Wc2 17:bc2 18:gc2 19:bec2 20:Wd 21:bd 22:Wl 23:bl
static void run_xconv(hipStream_t s, const float* pos, const float* xin, int Pp,
                      const StageCfg& cfg, const float* const* prm, float* out,
                      float* sq, int* knnIdx, float* rel, float* bufZ, float* bufZ2,
                      float* bufT1, float* bufT2, float* bufY,
                      float* stSum, float* stSq, float* stScale, float* stShift) {
  const int K = cfg.K, dil = cfg.dil, Cd = cfg.Cd, Cin = cfg.Cin, Cout = cfg.Cout;
  const int C = Cin + Cd;
  const int dm = (Cout + C - 1) / C;
  const int NP = BB * Pp;
  const int rows1 = NP * K;
  const int K2 = K * K;

  compute_sq<<<gs_grid(NP), 256, 0, s>>>(pos, sq, NP);
  knn_kernel<<<gs_grid(NP), 256, 0, s>>>(pos, sq, knnIdx, Pp, K * dil, dil, K);
  rel_kernel<<<gs_grid((long)rows1), 256, 0, s>>>(pos, knnIdx, rel, Pp, K);

  // lifted point features: rel@W1 -> spike -> bn ; h@W2 -> spike -> bn
  launch_gemm(s, rel, prm[0], prm[1], bufZ, rows1, Cd, 3);
  bn_spike(s, bufZ, rows1, Cd, prm[2], prm[3], stSum, stScale, stShift);
  launch_gemm(s, bufZ, prm[4], prm[5], bufZ2, rows1, Cd, Cd);
  bn_spike(s, bufZ2, rows1, Cd, prm[6], prm[7], stSum, stScale, stShift);

  // X-transform: pr@Wm -> spike -> bn ; Wc1 (+spike,bn) ; Wc2 (+bn)
  launch_gemm(s, rel, prm[8], prm[9], bufT1, NP, K2, 3 * K);
  bn_spike(s, bufT1, NP, K2, prm[10], prm[11], stSum, stScale, stShift);
  xform_conv<<<gs_grid((long)NP * K2), 256, 0, s>>>(bufT1, prm[12], prm[13], bufT2, NP, K, 1);
  bn_binary(s, bufT2, NP, K2, prm[14], prm[15], stSum, stScale, stShift);
  xform_conv<<<gs_grid((long)NP * K2), 256, 0, s>>>(bufT2, prm[16], prm[17], bufT1, NP, K, 0);
  bn_moments(s, bufT1, NP, K2, prm[18], prm[19], stSum, stSq, stScale, stShift);

  // xt = concat(h, gather(x)) @ t  ;  depthwise  (fused)
  depthwise_kernel<<<gs_grid((long)NP * C * dm), 256, 0, s>>>(
      bufZ2, xin, knnIdx, bufT1, prm[20], prm[21], bufY, NP, Pp, C, Cd, Cin, K, dm);

  // pointwise linear + stage spike
  launch_gemm(s, bufY, prm[22], prm[23], out, NP, Cout, C * dm);
  spike_inplace<<<gs_grid((long)NP * Cout), 256, 0, s>>>(out, (long)NP * Cout);
}

extern "C" void kernel_launch(void* const* d_in, const int* in_sizes, int n_in,
                              void* d_out, int out_size, void* d_ws, size_t ws_size,
                              hipStream_t stream) {
  (void)in_sizes; (void)n_in; (void)out_size; (void)ws_size;
  const float* pos0 = (const float*)d_in[0];   // [32*1024, 3]
  // d_in[1] = batch (unused: uniform cloud sizes)

  // params flattened in setup_inputs() dict insertion order:
  // [2..25]=c1, [26..49]=c2, [50..73]=c3, [74..97]=c4, 98..103 = Wl1,bl1,Wl2,bl2,Wl3,bl3
  const float* prm1[24]; const float* prm2[24]; const float* prm3[24]; const float* prm4[24];
  for (int i = 0; i < 24; i++) {
    prm1[i] = (const float*)d_in[2 + i];
    prm2[i] = (const float*)d_in[26 + i];
    prm3[i] = (const float*)d_in[50 + i];
    prm4[i] = (const float*)d_in[74 + i];
  }
  const float* Wl1 = (const float*)d_in[98];
  const float* bl1 = (const float*)d_in[99];
  const float* Wl2 = (const float*)d_in[100];
  const float* bl2 = (const float*)d_in[101];
  const float* Wl3 = (const float*)d_in[102];
  const float* bl3 = (const float*)d_in[103];

  // ---- workspace bump allocator (needs ~190 MB; buffers reused per stage) ----
  char* wsb = (char*)d_ws; size_t off = 0;
  auto alloc = [&](size_t bytes) -> void* {
    void* p = wsb + off; off += (bytes + 255) & ~(size_t)255; return p;
  };
  float* sq     = (float*)alloc((size_t)32768 * 4);
  int*   knnIdx = (int*)  alloc((size_t)262144 * 4);
  float* rel    = (float*)alloc((size_t)786432 * 4);
  float* bufZ   = (float*)alloc((size_t)16908288 * 4);   // max rows1*Cd (stage 4)
  float* bufZ2  = (float*)alloc((size_t)16908288 * 4);
  float* bufT1  = (float*)alloc((size_t)2097152 * 4);    // max NP*K*K (stage 1)
  float* bufT2  = (float*)alloc((size_t)2097152 * 4);
  float* bufY   = (float*)alloc((size_t)2097152 * 4);    // max NP*C*dm
  float* xA     = (float*)alloc((size_t)2097152 * 4);    // stage activations
  float* xB     = (float*)alloc((size_t)2097152 * 4);
  float* posB   = (float*)alloc((size_t)36864 * 4);      // [32,384,3]
  float* posC   = (float*)alloc((size_t)12384 * 4);      // [32,129,3]
  int*   fidx   = (int*)  alloc((size_t)(32 * 384) * 4);
  float* stSum  = (float*)alloc(256 * 4);
  float* stSq   = (float*)alloc(256 * 4);
  float* stScale= (float*)alloc(256 * 4);
  float* stShift= (float*)alloc(256 * 4);

  const StageCfg c1{0, 48, 32, 8, 1};
  const StageCfg c2{48, 96, 64, 12, 2};
  const StageCfg c3{96, 192, 128, 16, 2};
  const StageCfg c4{192, 384, 256, 16, 2};

  // stage 1 @ P=1024
  run_xconv(stream, pos0, nullptr, 1024, c1, prm1, xA,
            sq, knnIdx, rel, bufZ, bufZ2, bufT1, bufT2, bufY,
            stSum, stSq, stScale, stShift);

  // FPS 1024 -> 384, gather pos & x
  fps_kernel<<<BB, 256, 0, stream>>>(pos0, fidx, 1024, 384);
  gather_rows<<<gs_grid((long)BB * 384 * 3), 256, 0, stream>>>(pos0, fidx, posB, 1024, 384, 3);
  gather_rows<<<gs_grid((long)BB * 384 * 48), 256, 0, stream>>>(xA, fidx, xB, 1024, 384, 48);

  // stage 2 @ P=384
  run_xconv(stream, posB, xB, 384, c2, prm2, xA,
            sq, knnIdx, rel, bufZ, bufZ2, bufT1, bufT2, bufY,
            stSum, stSq, stScale, stShift);

  // FPS 384 -> 129, gather pos & x
  fps_kernel<<<BB, 256, 0, stream>>>(posB, fidx, 384, 129);
  gather_rows<<<gs_grid((long)BB * 129 * 3), 256, 0, stream>>>(posB, fidx, posC, 384, 129, 3);
  gather_rows<<<gs_grid((long)BB * 129 * 96), 256, 0, stream>>>(xA, fidx, xB, 384, 129, 96);

  // stage 3 @ P=129
  run_xconv(stream, posC, xB, 129, c3, prm3, xA,
            sq, knnIdx, rel, bufZ, bufZ2, bufT1, bufT2, bufY,
            stSum, stSq, stScale, stShift);

  // stage 4 @ P=129 (same positions)
  run_xconv(stream, posC, xA, 129, c4, prm4, xB,
            sq, knnIdx, rel, bufZ, bufZ2, bufT1, bufT2, bufY,
            stSum, stSq, stScale, stShift);

  // global mean pool + MLP head + log_softmax
  head_kernel<<<BB, 384, 0, stream>>>(xB, 129, Wl1, bl1, Wl2, bl2, Wl3, bl3,
                                      (float*)d_out);
}
